// GQAFractalAttention_30365418782766
// MI455X (gfx1250) — compile-verified, matrix operand order checked by reading, not run on hardware
//
#include <hip/hip_runtime.h>
#include <hip/hip_bf16.h>
#include <math.h>

#define DIMN 1024
#define SEQL 2048
#define QG 4
#define HD 64
#define FEAT 256          // heads-per-group * head_dim = 4*64
#define SCALE_F 0.125f    // 64^-0.5
#define LN_EPS 1e-5f

// GEMM macro-tile
#define BM 128
#define BN 128
#define BK 32
#define LDST 40           // padded LDS row stride in elements (64B data + 16B pad)

typedef __attribute__((ext_vector_type(16))) __bf16 v16bf;
typedef __attribute__((ext_vector_type(8)))  float  v8f;
typedef __attribute__((ext_vector_type(4)))  unsigned int u32x4;
typedef __attribute__((ext_vector_type(8)))  int i32x8;
typedef __attribute__((ext_vector_type(4)))  int i32x4;

#if defined(__has_builtin)
#if __has_builtin(__builtin_amdgcn_tensor_load_to_lds)
#define HAS_TDM 1
#endif
#endif
#ifndef HAS_TDM
#define HAS_TDM 0
#endif

union ATile { v16bf bf; unsigned int x[8]; unsigned short u[16]; };

__device__ __forceinline__ unsigned short f2bf(float f) {
  unsigned int u = __float_as_uint(f);
  unsigned int r = u + 0x7FFFu + ((u >> 16) & 1u);   // round-to-nearest-even
  return (unsigned short)(r >> 16);
}

// ---------------- elementwise helpers ----------------
__global__ void k_cvt_bf16(const float* __restrict__ in, unsigned short* __restrict__ out, int n) {
  int i = blockIdx.x * blockDim.x + threadIdx.x;
  if (i < n) out[i] = f2bf(in[i]);
}

__global__ void k_zero_f32(float* __restrict__ p, int n) {
  int i = blockIdx.x * blockDim.x + threadIdx.x;
  if (i < n) p[i] = 0.0f;
}

// split kv[L][2*DIM] -> K[G][L][FEAT] (row-major) and V^T[G][FEAT][L]
__global__ void k_split_kv(const unsigned short* __restrict__ kv,
                           unsigned short* __restrict__ kOut,
                           unsigned short* __restrict__ vT) {
  int i = blockIdx.x * blockDim.x + threadIdx.x;
  if (i >= SEQL * 2 * DIMN) return;
  int l    = i / (2 * DIMN);
  int col  = i % (2 * DIMN);
  int head = col >> 7;          // 128 cols per head (2*64)
  int pair = (col >> 6) & 1;    // 0 = k, 1 = v
  int d    = col & 63;
  int G    = head >> 2;
  int h    = head & 3;
  int f    = h * HD + d;
  unsigned short val = kv[i];
  if (pair == 0) kOut[((size_t)G * SEQL + l) * FEAT + f] = val;
  else           vT[((size_t)G * FEAT + f) * SEQL + l]   = val;
}

// ---------------- TDM: 2D tile (rows x 32 bf16) global -> LDS, padded rows ----------------
#if HAS_TDM
__device__ __forceinline__ void tdm_load_2d(unsigned short* ldsDst,
                                            const unsigned short* gsrc,
                                            int rows, int rowStrideElems) {
  unsigned lds_off = (unsigned)(unsigned long long)(uintptr_t)ldsDst; // low 32 bits = LDS offset
  unsigned long long ga = (unsigned long long)(uintptr_t)gsrc;
  u32x4 g0;
  g0[0] = 1u;                                        // count=1, user descriptor
  g0[1] = lds_off;                                   // lds_addr
  g0[2] = (unsigned)(ga & 0xFFFFFFFFull);            // global_addr[31:0]
  g0[3] = (unsigned)((ga >> 32) & 0x1FFFFFFull) | (2u << 30); // addr[56:32] | type=2
  unsigned td0 = 32u;                                // tensor/tile dim0 (elements per row)
  unsigned td1 = (unsigned)rows;
  unsigned long long st0 = (unsigned long long)rowStrideElems;
  i32x8 g1;
  // data_size=1 (2B) | pad_enable | pad_interval=3 (16 DW = 64B) | pad_amount=3 (4 DW = 16B)
  g1[0] = (int)((1u << 16) | (1u << 20) | (3u << 22) | (3u << 25));
  g1[1] = (int)((td0 & 0xFFFFu) << 16);              // tensor_dim0[15:0]
  g1[2] = (int)((td0 >> 16) | ((td1 & 0xFFFFu) << 16)); // dim0 hi | tensor_dim1[15:0]
  g1[3] = (int)((td1 >> 16) | (32u << 16));          // dim1 hi | tile_dim0=32
  g1[4] = (int)((unsigned)rows);                     // tile_dim1 (tile_dim2=0)
  g1[5] = (int)(st0 & 0xFFFFFFFFull);                // tensor_dim0_stride lo
  g1[6] = (int)((st0 >> 32) & 0xFFFFull);            // stride hi (tensor_dim1_stride=0)
  g1[7] = 0;
  i32x4 z4 = {0, 0, 0, 0};
#if __clang_major__ >= 23
  i32x8 z8 = {0, 0, 0, 0, 0, 0, 0, 0};
  __builtin_amdgcn_tensor_load_to_lds(g0, g1, z4, z4, z8, 0);
#else
  __builtin_amdgcn_tensor_load_to_lds(g0, g1, z4, z4, 0);
#endif
}
#endif

// ---------------- bf16 WMMA GEMM: C[M,N] = A[M,K] * B[N,K]^T ----------------
// 256 threads (8 waves) per block; 128x128 macro-tile; TDM double-buffered LDS staging.
// Wave w computes rows [bm0+16w, bm0+16w+16) x all 128 cols (8 WMMA tiles).
__global__ void k_gemm_bf16(const unsigned short* __restrict__ A,
                            const unsigned short* __restrict__ B,
                            int M, int N, int K,
                            float* __restrict__ Cf,
                            unsigned short* __restrict__ Cb,
                            const float* __restrict__ resid) {
  __shared__ unsigned short lA[2][BM * LDST];
  __shared__ unsigned short lB[2][BN * LDST];
  const int lane  = threadIdx.x & 31;
  const int wv    = __builtin_amdgcn_readfirstlane(threadIdx.x >> 5); // wave-uniform SGPR
  const int nBlkN = N / BN;
  const int bm0   = (blockIdx.x / nBlkN) * BM;
  const int bn0   = (blockIdx.x % nBlkN) * BN;
  const int nlo   = lane & 15, khalf = lane >> 4;

  v8f c[8];
  #pragma unroll
  for (int t = 0; t < 8; ++t)
    #pragma unroll
    for (int j = 0; j < 8; ++j) c[t][j] = 0.0f;

#if HAS_TDM
  if (wv == 0) {                                  // one wave drives the TDM (scalar branch)
    tdm_load_2d(&lA[0][0], A + (size_t)bm0 * K, BM, K);
    tdm_load_2d(&lB[0][0], B + (size_t)bn0 * K, BN, K);
  }
#endif

  for (int k0 = 0; k0 < K; k0 += BK) {
    const int cur = (k0 >> 5) & 1;
#if HAS_TDM
    if (wv == 0) {
      if (k0 + BK < K) {                          // prefetch next buffer, retire current
        tdm_load_2d(&lA[cur ^ 1][0], A + (size_t)bm0 * K + (k0 + BK), BM, K);
        tdm_load_2d(&lB[cur ^ 1][0], B + (size_t)bn0 * K + (k0 + BK), BN, K);
        __builtin_amdgcn_s_wait_tensorcnt(2);
      } else {
        __builtin_amdgcn_s_wait_tensorcnt(0);
      }
    }
    __syncthreads();
#else
    // cooperative fallback: all 256 threads copy both tiles (8B chunks, padded rows)
    for (int idx = threadIdx.x; idx < BM * 8; idx += 256) {
      int r = idx >> 3, cc = (idx & 7) * 4;
      *(unsigned long long*)(&lA[cur][r * LDST + cc]) =
          *(const unsigned long long*)(A + (size_t)(bm0 + r) * K + k0 + cc);
    }
    for (int idx = threadIdx.x; idx < BN * 8; idx += 256) {
      int r = idx >> 3, cc = (idx & 7) * 4;
      *(unsigned long long*)(&lB[cur][r * LDST + cc]) =
          *(const unsigned long long*)(B + (size_t)(bn0 + r) * K + k0 + cc);
    }
    __syncthreads();
#endif
    const unsigned short* la = &lA[cur][0];
    const unsigned short* lb = &lB[cur][0];
    // preload A tile + all 8 B tiles, then issue 8 back-to-back WMMAs
    ATile a, b[8];
    const int arow = wv * 16 + nlo;
    #pragma unroll
    for (int j = 0; j < 8; ++j) {
      int kk = ((j < 4) ? 0 : 16) + khalf * 8 + (j & 3) * 2;
      a.x[j] = *(const unsigned int*)(la + arow * LDST + kk);
    }
    #pragma unroll
    for (int t = 0; t < 8; ++t)
      #pragma unroll
      for (int j = 0; j < 8; ++j) {
        int kk = ((j < 4) ? 0 : 16) + khalf * 8 + (j & 3) * 2;
        b[t].x[j] = *(const unsigned int*)(lb + (t * 16 + nlo) * LDST + kk);
      }
    #pragma unroll
    for (int t = 0; t < 8; ++t)
      c[t] = __builtin_amdgcn_wmma_f32_16x16x32_bf16(false, a.bf, false, b[t].bf,
                                                     (short)0, c[t], false, false);
    __syncthreads();   // all reads of `cur` done before it is overwritten next iteration
  }

  #pragma unroll
  for (int t = 0; t < 8; ++t)
    #pragma unroll
    for (int j = 0; j < 8; ++j) {
      int row = bm0 + wv * 16 + j + khalf * 8;
      int col = bn0 + t * 16 + nlo;
      size_t idx = (size_t)row * N + col;
      float v = c[t][j];
      if (resid) v += resid[idx];
      if (Cf) Cf[idx] = v;
      if (Cb) Cb[idx] = f2bf(v);
    }
}

// ---------------- fractal GQA attention ----------------
// One wave per (16 query rows, query-group g). out_g += sum_G softmax(Q_g K_G^T * s) V_G
__global__ void k_attn(const unsigned short* __restrict__ qb,   // [L][DIM] bf16
                       const unsigned short* __restrict__ kb,   // [G][L][FEAT]
                       const unsigned short* __restrict__ vT,   // [G][FEAT][L]
                       float* __restrict__ outAcc) {            // [L][DIM] f32, pre-zeroed
  __shared__ unsigned short pLds[16 * 32];
  const int lane  = threadIdx.x;
  const int nlo   = lane & 15, khalf = lane >> 4;
  const int mt    = blockIdx.x % (SEQL / 16);
  const int g     = blockIdx.x / (SEQL / 16);
  const int m0    = mt * 16;

  // Q: 16 rows x 256 features -> 8 A-tiles (16x32), resident in registers
  ATile aQ[8];
  #pragma unroll
  for (int t = 0; t < 8; ++t)
    #pragma unroll
    for (int j = 0; j < 8; ++j) {
      int kf = t * 32 + ((j < 4) ? 0 : 16) + khalf * 8 + (j & 3) * 2;
      aQ[t].x[j] = *(const unsigned int*)(qb + (size_t)(m0 + nlo) * DIMN + g * FEAT + kf);
    }

  for (int G = 0; G < QG; ++G) {
    const unsigned short* kG = kb + (size_t)G * SEQL * FEAT;
    const unsigned short* vG = vT + (size_t)G * FEAT * SEQL;
    float mrun[8], lrun[8];
    #pragma unroll
    for (int j = 0; j < 8; ++j) { mrun[j] = -INFINITY; lrun[j] = 0.0f; }
    v8f o[16];
    #pragma unroll
    for (int t = 0; t < 16; ++t)
      #pragma unroll
      for (int j = 0; j < 8; ++j) o[t][j] = 0.0f;

    for (int kt = 0; kt < SEQL; kt += 32) {
      // prefetch next key chunk into cache while this one computes
      if (kt + 32 < SEQL) {
        __builtin_prefetch(kG + (size_t)(kt + 32 + nlo) * FEAT, 0, 1);
        __builtin_prefetch(kG + (size_t)(kt + 48 + nlo) * FEAT, 0, 1);
      }
      // ---- scores S = Q K^T for 32 keys: two 16x16 tiles, K-dim 256 ----
      v8f s0, s1;
      #pragma unroll
      for (int j = 0; j < 8; ++j) { s0[j] = 0.0f; s1[j] = 0.0f; }
      #pragma unroll
      for (int t2 = 0; t2 < 8; t2 += 2) {          // preload 4 K tiles, then 4 WMMAs
        ATile kb4[4];
        #pragma unroll
        for (int u = 0; u < 2; ++u)
          #pragma unroll
          for (int j = 0; j < 8; ++j) {
            int kf = (t2 + u) * 32 + ((j < 4) ? 0 : 16) + khalf * 8 + (j & 3) * 2;
            kb4[u * 2 + 0].x[j] = *(const unsigned int*)(kG + (size_t)(kt + nlo) * FEAT + kf);
            kb4[u * 2 + 1].x[j] = *(const unsigned int*)(kG + (size_t)(kt + 16 + nlo) * FEAT + kf);
          }
        #pragma unroll
        for (int u = 0; u < 2; ++u) {
          s0 = __builtin_amdgcn_wmma_f32_16x16x32_bf16(false, aQ[t2 + u].bf, false,
                                                       kb4[u * 2 + 0].bf, (short)0, s0, false, false);
          s1 = __builtin_amdgcn_wmma_f32_16x16x32_bf16(false, aQ[t2 + u].bf, false,
                                                       kb4[u * 2 + 1].bf, (short)0, s1, false, false);
        }
      }
      // ---- online softmax (rows j+8*khalf; columns across the 16-lane group) ----
      float rm[8], rs[8];
      #pragma unroll
      for (int j = 0; j < 8; ++j) {
        float a = s0[j] * SCALE_F, b2 = s1[j] * SCALE_F;
        s0[j] = a; s1[j] = b2;
        rm[j] = fmaxf(a, b2);
      }
      #pragma unroll
      for (int off = 1; off < 16; off <<= 1)
        #pragma unroll
        for (int j = 0; j < 8; ++j)
          rm[j] = fmaxf(rm[j], __shfl_xor(rm[j], off, 32));
      float scl[8];
      #pragma unroll
      for (int j = 0; j < 8; ++j) {
        float mnew = fmaxf(mrun[j], rm[j]);
        scl[j]  = __expf(mrun[j] - mnew);
        mrun[j] = mnew;
        float p0 = __expf(s0[j] - mnew), p1 = __expf(s1[j] - mnew);
        s0[j] = p0; s1[j] = p1;
        rs[j] = p0 + p1;
      }
      #pragma unroll
      for (int off = 1; off < 16; off <<= 1)
        #pragma unroll
        for (int j = 0; j < 8; ++j)
          rs[j] += __shfl_xor(rs[j], off, 32);
      #pragma unroll
      for (int j = 0; j < 8; ++j) lrun[j] = lrun[j] * scl[j] + rs[j];
      #pragma unroll
      for (int t = 0; t < 16; ++t)
        #pragma unroll
        for (int j = 0; j < 8; ++j) o[t][j] *= scl[j];

      // ---- P (D-layout) -> LDS -> A-layout bf16 ----
      __syncthreads();
      #pragma unroll
      for (int j = 0; j < 8; ++j) {
        int row = j + khalf * 8;
        pLds[row * 32 + nlo]      = f2bf(s0[j]);
        pLds[row * 32 + 16 + nlo] = f2bf(s1[j]);
      }
      __syncthreads();
      ATile aP;
      #pragma unroll
      for (int j = 0; j < 8; ++j) {
        int kf = ((j < 4) ? 0 : 16) + khalf * 8 + (j & 3) * 2;
        aP.x[j] = *(const unsigned int*)(pLds + nlo * 32 + kf);
      }
      // ---- O += P * V : 16 feature tiles, V preloaded in groups of 4 ----
      #pragma unroll
      for (int f4 = 0; f4 < 16; f4 += 4) {
        ATile vb4[4];
        #pragma unroll
        for (int u = 0; u < 4; ++u)
          #pragma unroll
          for (int j = 0; j < 8; ++j) {
            int kf = ((j < 4) ? 0 : 16) + khalf * 8 + (j & 3) * 2;
            vb4[u].x[j] = *(const unsigned int*)(vG + (size_t)((f4 + u) * 16 + nlo) * SEQL + kt + kf);
          }
        #pragma unroll
        for (int u = 0; u < 4; ++u)
          o[f4 + u] = __builtin_amdgcn_wmma_f32_16x16x32_bf16(false, aP.bf, false, vb4[u].bf,
                                                              (short)0, o[f4 + u], false, false);
      }
    }
    // ---- accumulate this G's normalized output ----
    float linv[8];
    #pragma unroll
    for (int j = 0; j < 8; ++j) linv[j] = 1.0f / lrun[j];
    #pragma unroll
    for (int ft = 0; ft < 16; ++ft)
      #pragma unroll
      for (int j = 0; j < 8; ++j) {
        int row = m0 + j + khalf * 8;
        int col = g * FEAT + ft * 16 + nlo;
        outAcc[(size_t)row * DIMN + col] += o[ft][j] * linv[j];
      }
  }
}

// ---------------- LayerNorm (one block per row) ----------------
__global__ void k_layernorm(const float* __restrict__ y, const float* __restrict__ gamma,
                            const float* __restrict__ beta, float* __restrict__ out) {
  __shared__ float red[256];
  int row = blockIdx.x;
  const float* yr = y + (size_t)row * DIMN;
  float s = 0.0f;
  for (int i = threadIdx.x; i < DIMN; i += blockDim.x) s += yr[i];
  red[threadIdx.x] = s; __syncthreads();
  for (int off = 128; off > 0; off >>= 1) {
    if (threadIdx.x < off) red[threadIdx.x] += red[threadIdx.x + off];
    __syncthreads();
  }
  float mu = red[0] * (1.0f / DIMN);
  __syncthreads();
  float v = 0.0f;
  for (int i = threadIdx.x; i < DIMN; i += blockDim.x) { float d = yr[i] - mu; v += d * d; }
  red[threadIdx.x] = v; __syncthreads();
  for (int off = 128; off > 0; off >>= 1) {
    if (threadIdx.x < off) red[threadIdx.x] += red[threadIdx.x + off];
    __syncthreads();
  }
  float rstd = rsqrtf(red[0] * (1.0f / DIMN) + LN_EPS);
  for (int i = threadIdx.x; i < DIMN; i += blockDim.x)
    out[(size_t)row * DIMN + i] = gamma[i] * (yr[i] - mu) * rstd + beta[i];
}

// ---------------- host orchestration ----------------
extern "C" void kernel_launch(void* const* d_in, const int* in_sizes, int n_in,
                              void* d_out, int out_size, void* d_ws, size_t ws_size,
                              hipStream_t stream) {
  (void)in_sizes; (void)n_in; (void)out_size; (void)ws_size;
  const float* x     = (const float*)d_in[0];
  const float* Wq    = (const float*)d_in[1];
  const float* Wkv   = (const float*)d_in[2];
  const float* Wo    = (const float*)d_in[3];
  const float* gamma = (const float*)d_in[4];
  const float* beta  = (const float*)d_in[5];
  float* out = (float*)d_out;

  char* ws = (char*)d_ws;
  size_t off = 0;
  auto alloc = [&](size_t bytes) -> void* {
    void* p = ws + off; off += (bytes + 255) & ~(size_t)255; return p;
  };
  unsigned short* xb    = (unsigned short*)alloc((size_t)SEQL * DIMN * 2);
  unsigned short* wqb   = (unsigned short*)alloc((size_t)DIMN * DIMN * 2);
  unsigned short* wkvb  = (unsigned short*)alloc((size_t)2 * DIMN * DIMN * 2);
  unsigned short* wob   = (unsigned short*)alloc((size_t)DIMN * DIMN * 2);
  unsigned short* qb    = (unsigned short*)alloc((size_t)SEQL * DIMN * 2);
  unsigned short* kvb   = (unsigned short*)alloc((size_t)SEQL * 2 * DIMN * 2);
  unsigned short* kb    = (unsigned short*)alloc((size_t)QG * SEQL * FEAT * 2);
  unsigned short* vTb   = (unsigned short*)alloc((size_t)QG * FEAT * SEQL * 2);
  float*          attnF = (float*)alloc((size_t)SEQL * DIMN * 4);
  unsigned short* attnB = (unsigned short*)alloc((size_t)SEQL * DIMN * 2);
  float*          yF    = (float*)alloc((size_t)SEQL * DIMN * 4);

  const int T = 256;
  // bf16 conversions
  k_cvt_bf16<<<(SEQL * DIMN + T - 1) / T, T, 0, stream>>>(x, xb, SEQL * DIMN);
  k_cvt_bf16<<<(DIMN * DIMN + T - 1) / T, T, 0, stream>>>(Wq, wqb, DIMN * DIMN);
  k_cvt_bf16<<<(2 * DIMN * DIMN + T - 1) / T, T, 0, stream>>>(Wkv, wkvb, 2 * DIMN * DIMN);
  k_cvt_bf16<<<(DIMN * DIMN + T - 1) / T, T, 0, stream>>>(Wo, wob, DIMN * DIMN);

  // q = x Wq^T ; kv = x Wkv^T  (128x128 macro-tiles, 8 waves/block)
  k_gemm_bf16<<<(SEQL / BM) * (DIMN / BN), T, 0, stream>>>(xb, wqb, SEQL, DIMN, DIMN,
                                                           nullptr, qb, nullptr);
  k_gemm_bf16<<<(SEQL / BM) * (2 * DIMN / BN), T, 0, stream>>>(xb, wkvb, SEQL, 2 * DIMN, DIMN,
                                                               nullptr, kvb, nullptr);
  k_split_kv<<<(SEQL * 2 * DIMN + T - 1) / T, T, 0, stream>>>(kvb, kb, vTb);

  // attention
  k_zero_f32<<<(SEQL * DIMN + T - 1) / T, T, 0, stream>>>(attnF, SEQL * DIMN);
  k_attn<<<(SEQL / 16) * QG, 32, 0, stream>>>(qb, kb, vTb, attnF);
  k_cvt_bf16<<<(SEQL * DIMN + T - 1) / T, T, 0, stream>>>(attnF, attnB, SEQL * DIMN);

  // y = x + attn Wo^T ; then LayerNorm
  k_gemm_bf16<<<(SEQL / BM) * (DIMN / BN), T, 0, stream>>>(attnB, wob, SEQL, DIMN, DIMN,
                                                           yF, nullptr, x);
  k_layernorm<<<SEQL, T, 0, stream>>>(yF, gamma, beta, out);
}
